// CausalSelfAttention_18674517803607
// MI455X (gfx1250) — matrix operand order repaired and checked by reading, not compiled, not run on hardware
//
#include <hip/hip_runtime.h>

#define TT 2048
#define CC 1024
#define HH 16
#define DD 64
#define BB 4
#define MM (BB*TT)      // 8192
#define NQKV (3*CC)     // 3072

typedef __attribute__((ext_vector_type(16))) __bf16 v16bf;
typedef __attribute__((ext_vector_type(8)))  __bf16 v8bf;
typedef __attribute__((ext_vector_type(8)))  float  v8f;

// A-fragment (16x32 bf16): lane holds row lane%16; K contiguous [16h..16h+15]
static __device__ __forceinline__ v16bf load_a_frag(const __bf16* row, int half) {
    v8bf lo = *(const v8bf*)(row + 16 * half);
    v8bf hi = *(const v8bf*)(row + 16 * half + 8);
    v16bf a;
#pragma unroll
    for (int i = 0; i < 8; ++i) { a[i] = lo[i]; a[i + 8] = hi[i]; }
    return a;
}

__global__ void cvt_f32_bf16(const float* __restrict__ src, __bf16* __restrict__ dst, int n) {
    int i = blockIdx.x * blockDim.x + threadIdx.x;
    int stride = gridDim.x * blockDim.x;
    for (; i < n; i += stride) dst[i] = (__bf16)src[i];
}

// ---------------------------------------------------------------------------
// qkv = x @ Wqkv^T + b ; split into Q[B,H,T,D], K[B,H,T,D], Vt[B,H,D,T] (bf16)
// 64x64 per wave (4x4 WMMA register blocking), 8 waves/block -> 128x256/block
// ---------------------------------------------------------------------------
__global__ __launch_bounds__(256)
void qkv_gemm(const __bf16* __restrict__ xb, const __bf16* __restrict__ wb,
              const float* __restrict__ bias,
              __bf16* __restrict__ qb, __bf16* __restrict__ kb, __bf16* __restrict__ vtb) {
    const int lane = threadIdx.x & 31;
    const int wave = threadIdx.x >> 5;
    const int l16  = lane & 15;
    const int half = lane >> 4;
    const int mBase = blockIdx.x * 128 + (wave >> 2) * 64;
    const int nBase = blockIdx.y * 256 + (wave & 3) * 64;

    const __bf16* aBase = xb + (size_t)(mBase + l16) * CC;
    const __bf16* bBase = wb + (size_t)(nBase + l16) * CC;

    v8f acc[4][4];
#pragma unroll
    for (int mt = 0; mt < 4; ++mt)
#pragma unroll
        for (int nt = 0; nt < 4; ++nt) acc[mt][nt] = (v8f){};

    for (int k0 = 0; k0 < CC; k0 += 32) {
        v16bf aF[4], bF[4];
#pragma unroll
        for (int mt = 0; mt < 4; ++mt) aF[mt] = load_a_frag(aBase + mt * 16 * CC + k0, half);
#pragma unroll
        for (int nt = 0; nt < 4; ++nt) bF[nt] = *(const v16bf*)(bBase + nt * 16 * CC + k0 + 16 * half);
#pragma unroll
        for (int mt = 0; mt < 4; ++mt)
#pragma unroll
            for (int nt = 0; nt < 4; ++nt)
                acc[mt][nt] = __builtin_amdgcn_wmma_f32_16x16x32_bf16(
                    false, aF[mt], false, bF[nt], (short)0, acc[mt][nt], false, false);
    }

#pragma unroll
    for (int nt = 0; nt < 4; ++nt) {
        const int n   = nBase + nt * 16 + l16;
        const float bv = bias[n];
        const int sec  = n >> 10;          // 0=q 1=k 2=v
        const int nn   = n & (CC - 1);
        const int head = nn >> 6;
        const int d    = nn & 63;
#pragma unroll
        for (int mt = 0; mt < 4; ++mt)
#pragma unroll
            for (int r = 0; r < 8; ++r) {
                const int m  = mBase + mt * 16 + r + 8 * half;
                const int b_ = m >> 11;
                const int t  = m & (TT - 1);
                const int bh = b_ * HH + head;
                const float v = acc[mt][nt][r] + bv;
                if (sec == 0)      qb[((size_t)bh * TT + t) * DD + d]  = (__bf16)v;
                else if (sec == 1) kb[((size_t)bh * TT + t) * DD + d]  = (__bf16)v;
                else               vtb[((size_t)bh * DD + d) * TT + t] = (__bf16)v;
            }
    }
}

// ---------------------------------------------------------------------------
// Causal flash attention: one wave per (b, h, 32-row Q block); 32-key steps.
// K/V fragments shared across the two 16-row Q tiles.
// ---------------------------------------------------------------------------
__global__ __launch_bounds__(32)
void flash_attn(const __bf16* __restrict__ qb, const __bf16* __restrict__ kb,
                const __bf16* __restrict__ vtb, __bf16* __restrict__ attb) {
    __shared__ __align__(32) __bf16 ldsP[32 * 32];
    const int lane  = threadIdx.x & 31;
    const int l16   = lane & 15;
    const int half  = lane >> 4;
    const int qBase = blockIdx.x * 32;
    const int bh    = blockIdx.y;
    const int b_    = bh >> 4;
    const int h     = bh & 15;

    const __bf16* Q  = qb  + (size_t)bh * TT * DD;
    const __bf16* K  = kb  + (size_t)bh * TT * DD;
    const __bf16* Vt = vtb + (size_t)bh * DD * TT;

    v16bf aQ[2][2];
#pragma unroll
    for (int qt = 0; qt < 2; ++qt) {
        const __bf16* qrow = Q + (size_t)(qBase + qt * 16 + l16) * DD;
        aQ[qt][0] = load_a_frag(qrow, half);        // d 0..31
        aQ[qt][1] = load_a_frag(qrow + 32, half);   // d 32..63
    }

    v8f m_run[2], l_run[2], O[2][4];
#pragma unroll
    for (int qt = 0; qt < 2; ++qt) {
#pragma unroll
        for (int r = 0; r < 8; ++r) { m_run[qt][r] = -3.0e38f; l_run[qt][r] = 0.f; }
#pragma unroll
        for (int dt = 0; dt < 4; ++dt) O[qt][dt] = (v8f){};
    }

    const int kbCount = blockIdx.x + 1;    // keys 0 .. qBase+31
    for (int j = 0; j < kbCount; ++j) {
        const int key0 = j * 32;
        if (j + 1 < kbCount) {             // prefetch next K/V block
            __builtin_prefetch(K + (size_t)(key0 + 32 + l16) * DD, 0, 3);
            __builtin_prefetch(Vt + (size_t)l16 * TT + key0 + 32, 0, 3);
        }
        // K B-fragments: [key-half][k-step]; B[k=d][n=key] = K[key][d]
        v16bf bK[2][2];
#pragma unroll
        for (int kh = 0; kh < 2; ++kh) {
            const __bf16* kr = K + (size_t)(key0 + l16) * DD + kh * 16 * DD + 16 * half;
            bK[kh][0] = *(const v16bf*)(kr);
            bK[kh][1] = *(const v16bf*)(kr + 32);
        }
        v8f S[2][2] = {};
#pragma unroll
        for (int qt = 0; qt < 2; ++qt)
#pragma unroll
            for (int kh = 0; kh < 2; ++kh) {
                S[qt][kh] = __builtin_amdgcn_wmma_f32_16x16x32_bf16(
                    false, aQ[qt][0], false, bK[kh][0], (short)0, S[qt][kh], false, false);
                S[qt][kh] = __builtin_amdgcn_wmma_f32_16x16x32_bf16(
                    false, aQ[qt][1], false, bK[kh][1], (short)0, S[qt][kh], false, false);
            }

        const float NEG = -3.0e38f;
        const int key_n0 = key0 + l16;
        const int key_n1 = key0 + 16 + l16;
#pragma unroll
        for (int qt = 0; qt < 2; ++qt) {
            v8f m_new, alpha, P0, P1, rsum;
#pragma unroll
            for (int r = 0; r < 8; ++r) {
                const int qg = qBase + qt * 16 + r + 8 * half;
                float s0 = (key_n0 > qg) ? NEG : S[qt][0][r] * 0.125f;   // 1/sqrt(64)
                float s1 = (key_n1 > qg) ? NEG : S[qt][1][r] * 0.125f;
                float t = fmaxf(s0, s1);                                 // row max (16 lanes)
                t = fmaxf(t, __shfl_xor(t, 1, 32));
                t = fmaxf(t, __shfl_xor(t, 2, 32));
                t = fmaxf(t, __shfl_xor(t, 4, 32));
                t = fmaxf(t, __shfl_xor(t, 8, 32));
                const float mn = fmaxf(m_run[qt][r], t);
                m_new[r] = mn;
                alpha[r] = __expf(m_run[qt][r] - mn);
                const float p0 = __expf(s0 - mn), p1 = __expf(s1 - mn);
                P0[r] = p0; P1[r] = p1;
                float s = p0 + p1;
                s += __shfl_xor(s, 1, 32);
                s += __shfl_xor(s, 2, 32);
                s += __shfl_xor(s, 4, 32);
                s += __shfl_xor(s, 8, 32);
                rsum[r] = s;
            }
#pragma unroll
            for (int r = 0; r < 8; ++r) {
                l_run[qt][r] = l_run[qt][r] * alpha[r] + rsum[r];
                m_run[qt][r] = m_new[r];
                O[qt][0][r] *= alpha[r]; O[qt][1][r] *= alpha[r];
                O[qt][2][r] *= alpha[r]; O[qt][3][r] *= alpha[r];
            }
            // C-layout P -> A-layout staging tile (rows qt*16 .. qt*16+15)
#pragma unroll
            for (int r = 0; r < 8; ++r) {
                const int row = qt * 16 + r + 8 * half;
                ldsP[row * 32 + l16]      = (__bf16)P0[r];
                ldsP[row * 32 + 16 + l16] = (__bf16)P1[r];
            }
        }
        asm volatile("s_wait_dscnt 0" ::: "memory");
        v16bf aP[2];
#pragma unroll
        for (int qt = 0; qt < 2; ++qt)
            aP[qt] = load_a_frag(&ldsP[(qt * 16 + l16) * 32], half);
        // O += P * V : B[k=key][n=d] = Vt[d][key] (contiguous along T)
#pragma unroll
        for (int dt = 0; dt < 4; ++dt) {
            v16bf bv = *(const v16bf*)(Vt + (size_t)(dt * 16 + l16) * TT + key0 + 16 * half);
#pragma unroll
            for (int qt = 0; qt < 2; ++qt)
                O[qt][dt] = __builtin_amdgcn_wmma_f32_16x16x32_bf16(
                    false, aP[qt], false, bv, (short)0, O[qt][dt], false, false);
        }
        asm volatile("s_wait_dscnt 0" ::: "memory");   // reads done before next block's writes
    }

#pragma unroll
    for (int qt = 0; qt < 2; ++qt) {
        float inv[8];
#pragma unroll
        for (int r = 0; r < 8; ++r) inv[r] = 1.0f / l_run[qt][r];
#pragma unroll
        for (int dt = 0; dt < 4; ++dt)
#pragma unroll
            for (int r = 0; r < 8; ++r) {
                const int t = qBase + qt * 16 + r + 8 * half;
                attb[((size_t)(b_ * TT + t)) * CC + h * 64 + dt * 16 + l16] =
                    (__bf16)(O[qt][dt][r] * inv[r]);
            }
    }
}

// ---------------------------------------------------------------------------
// out = attb @ Wout^T + b_out  (fp32 result), 64x64 per wave
// ---------------------------------------------------------------------------
__global__ __launch_bounds__(256)
void out_gemm(const __bf16* __restrict__ ab, const __bf16* __restrict__ wb,
              const float* __restrict__ bias, float* __restrict__ out) {
    const int lane = threadIdx.x & 31;
    const int wave = threadIdx.x >> 5;
    const int l16  = lane & 15;
    const int half = lane >> 4;
    const int mBase = blockIdx.x * 128 + (wave >> 2) * 64;
    const int nBase = blockIdx.y * 256 + (wave & 3) * 64;

    const __bf16* aBase = ab + (size_t)(mBase + l16) * CC;
    const __bf16* bBase = wb + (size_t)(nBase + l16) * CC;

    v8f acc[4][4];
#pragma unroll
    for (int mt = 0; mt < 4; ++mt)
#pragma unroll
        for (int nt = 0; nt < 4; ++nt) acc[mt][nt] = (v8f){};

    for (int k0 = 0; k0 < CC; k0 += 32) {
        v16bf aF[4], bF[4];
#pragma unroll
        for (int mt = 0; mt < 4; ++mt) aF[mt] = load_a_frag(aBase + mt * 16 * CC + k0, half);
#pragma unroll
        for (int nt = 0; nt < 4; ++nt) bF[nt] = *(const v16bf*)(bBase + nt * 16 * CC + k0 + 16 * half);
#pragma unroll
        for (int mt = 0; mt < 4; ++mt)
#pragma unroll
            for (int nt = 0; nt < 4; ++nt)
                acc[mt][nt] = __builtin_amdgcn_wmma_f32_16x16x32_bf16(
                    false, aF[mt], false, bF[nt], (short)0, acc[mt][nt], false, false);
    }

#pragma unroll
    for (int nt = 0; nt < 4; ++nt) {
        const int n = nBase + nt * 16 + l16;
        const float bv = bias[n];
#pragma unroll
        for (int mt = 0; mt < 4; ++mt)
#pragma unroll
            for (int r = 0; r < 8; ++r) {
                const int m = mBase + mt * 16 + r + 8 * half;
                out[(size_t)m * CC + n] = acc[mt][nt][r] + bv;
            }
    }
}

extern "C" void kernel_launch(void* const* d_in, const int* in_sizes, int n_in,
                              void* d_out, int out_size, void* d_ws, size_t ws_size,
                              hipStream_t stream) {
    const float* x     = (const float*)d_in[0];
    const float* w_qkv = (const float*)d_in[1];
    const float* b_qkv = (const float*)d_in[2];
    const float* w_out = (const float*)d_in[3];
    const float* b_out = (const float*)d_in[4];

    char* ws = (char*)d_ws;
    auto alloc = [&](size_t bytes) {
        char* p = ws;
        ws += (bytes + 255) & ~(size_t)255;
        return p;
    };
    __bf16* xb    = (__bf16*)alloc((size_t)MM * CC * 2);
    __bf16* wqkvb = (__bf16*)alloc((size_t)NQKV * CC * 2);
    __bf16* woutb = (__bf16*)alloc((size_t)CC * CC * 2);
    __bf16* qb    = (__bf16*)alloc((size_t)BB * HH * TT * DD * 2);
    __bf16* kb    = (__bf16*)alloc((size_t)BB * HH * TT * DD * 2);
    __bf16* vtb   = (__bf16*)alloc((size_t)BB * HH * DD * TT * 2);
    __bf16* attb  = (__bf16*)alloc((size_t)MM * CC * 2);

    cvt_f32_bf16<<<2048, 256, 0, stream>>>(x, xb, MM * CC);
    cvt_f32_bf16<<<2048, 256, 0, stream>>>(w_qkv, wqkvb, NQKV * CC);
    cvt_f32_bf16<<<1024, 256, 0, stream>>>(w_out, woutb, CC * CC);

    qkv_gemm<<<dim3(MM / 128, NQKV / 256), 256, 0, stream>>>(xb, wqkvb, b_qkv, qb, kb, vtb);
    flash_attn<<<dim3(TT / 32, BB * HH), 32, 0, stream>>>(qb, kb, vtb, attb);
    out_gemm<<<dim3(MM / 128, CC / 256), 256, 0, stream>>>(attb, woutb, b_out, (float*)d_out);
}
